// Basic_Block_37726992728103
// MI455X (gfx1250) — matrix operand order repaired and checked by reading, not compiled, not run on hardware
//
#include <hip/hip_runtime.h>
#include <hip/hip_bf16.h>

// ---------------------------------------------------------------------------
// Mamba block for MI455X (gfx1250, wave32, WMMA).
// All four projections run through v_wmma_f32_16x16x32_bf16 (f32 accum) with
// MTxNT register blocking per wave for fragment reuse.
// ---------------------------------------------------------------------------

typedef __attribute__((ext_vector_type(16))) __bf16       v16bf;
typedef __attribute__((ext_vector_type(8)))  float        v8f;
typedef __attribute__((ext_vector_type(8)))  unsigned int u32x8;

constexpr int   Bb   = 4;
constexpr int   Ll   = 1024;
constexpr int   Dd   = 768;
constexpr int   DIc  = 1536;           // d_inner
constexpr int   Nst  = 16;             // d_state
constexpr int   DTRc = 48;             // dt_rank
constexpr int   KDT  = 64;             // dt_rank padded to WMMA K granularity
constexpr int   NXc  = DTRc + 2*Nst;   // 80
constexpr int   TOK  = Bb * Ll;        // 4096 tokens
constexpr float EPSc = 1e-6f;

__device__ __forceinline__ __bf16 f2bf(float f) { return (__bf16)f; }

__device__ __forceinline__ float softplusf(float x) {
    return (x > 20.f) ? x : log1pf(expf(x));
}

// ---------------------------------------------------------------------------
// WMMA fragment loads (layouts per CDNA5 ISA 7.12.2).
//   A 16x32: lane(0-15)=row m, K {0-7,16-23}; lane(16-31)=row m, K {8-15,24-31}
//   B 32x16: lane(0-15)=col n, K {0-7,16-23}; lane(16-31)=col n, K {8-15,24-31}
// Both operands are K-contiguous in memory, so each frag is 2x b128 per lane.
// ---------------------------------------------------------------------------
__device__ __forceinline__ v16bf load_a_frag(const __bf16* row, int kk, int kh) {
    uint4 lo = *(const uint4*)(row + kk + kh * 8);
    uint4 hi = *(const uint4*)(row + kk + 16 + kh * 8);
    u32x8 t;
    t[0]=lo.x; t[1]=lo.y; t[2]=lo.z; t[3]=lo.w;
    t[4]=hi.x; t[5]=hi.y; t[6]=hi.z; t[7]=hi.w;
    return __builtin_bit_cast(v16bf, t);
}
__device__ __forceinline__ v16bf load_b_frag(const __bf16* row, int kk, int kh) {
    uint4 lo = *(const uint4*)(row + kk + kh * 8);
    uint4 hi = *(const uint4*)(row + kk + 16 + kh * 8);
    u32x8 t;
    t[0]=lo.x; t[1]=lo.y; t[2]=lo.z; t[3]=lo.w;
    t[4]=hi.x; t[5]=hi.y; t[6]=hi.z; t[7]=hi.w;
    return __builtin_bit_cast(v16bf, t);
}

// ---------------------------------------------------------------------------
// WMMA GEMM: C[M,N] = A[M,K] * W[N,K]^T, bf16 inputs, f32 accumulate.
// Each wave owns an MT x NT group of 16x16 tiles: per k-step it loads MT A
// frags + NT B frags and issues MT*NT WMMAs (fragment reuse in registers).
// C/D layout: VGPR i -> row i (lanes 0-15) / row i+8 (lanes 16-31), col=lane&15
// ---------------------------------------------------------------------------
template<int MT, int NT>
__global__ void gemm_bf16_wmma_kernel(const __bf16* __restrict__ A, int lda,
                                      const __bf16* __restrict__ W, int ldw,
                                      float* __restrict__ C, int ldc,
                                      int M, int N, int K) {
    const int wave = blockIdx.x * (blockDim.x >> 5) + (threadIdx.x >> 5);
    const int grpN = N / (16 * NT);
    const int grpM = M / (16 * MT);
    if (wave >= grpM * grpN) return;            // wave-uniform: EXEC stays full

    const int m0 = (wave / grpN) * (16 * MT);
    const int n0 = (wave % grpN) * (16 * NT);
    const int lane = threadIdx.x & 31;
    const int r  = lane & 15;                   // A row / output col / W row
    const int kh = lane >> 4;                   // K-half select

    const __bf16* Arow[MT];
#pragma unroll
    for (int mi = 0; mi < MT; ++mi)
        Arow[mi] = A + (size_t)(m0 + mi * 16 + r) * lda;
    const __bf16* Wrow[NT];
#pragma unroll
    for (int ni = 0; ni < NT; ++ni)
        Wrow[ni] = W + (size_t)(n0 + ni * 16 + r) * ldw;

    const v8f zero = {};
    v8f acc[MT][NT];
#pragma unroll
    for (int mi = 0; mi < MT; ++mi)
#pragma unroll
        for (int ni = 0; ni < NT; ++ni) acc[mi][ni] = zero;

    for (int kk = 0; kk < K; kk += 32) {
#pragma unroll
        for (int mi = 0; mi < MT; ++mi)
            __builtin_prefetch(Arow[mi] + kk + 64, 0, 0);  // global_prefetch_b8
#pragma unroll
        for (int ni = 0; ni < NT; ++ni)
            __builtin_prefetch(Wrow[ni] + kk + 64, 0, 0);

        v16bf av[MT];
#pragma unroll
        for (int mi = 0; mi < MT; ++mi) av[mi] = load_a_frag(Arow[mi], kk, kh);
#pragma unroll
        for (int ni = 0; ni < NT; ++ni) {
            v16bf bv = load_b_frag(Wrow[ni], kk, kh);
#pragma unroll
            for (int mi = 0; mi < MT; ++mi)
                acc[mi][ni] = __builtin_amdgcn_wmma_f32_16x16x32_bf16(
                    false, av[mi], false, bv, (short)0, acc[mi][ni],
                    false, false);
        }
    }
#pragma unroll
    for (int mi = 0; mi < MT; ++mi)
#pragma unroll
        for (int ni = 0; ni < NT; ++ni)
#pragma unroll
            for (int i = 0; i < 8; ++i)
                C[(size_t)(m0 + mi * 16 + i + kh * 8) * ldc
                  + n0 + ni * 16 + r] = acc[mi][ni][i];
}

// ---------------------------------------------------------------------------
// Generic fp32 -> bf16 conversion
// ---------------------------------------------------------------------------
__global__ void cvt_f32_bf16_kernel(const float* __restrict__ src,
                                    __bf16* __restrict__ dst, int n) {
    int i = blockIdx.x * blockDim.x + threadIdx.x;
    if (i < n) dst[i] = f2bf(src[i]);
}

// W_dt (DI,48) -> bf16 (DI,64) zero-padded along K
__global__ void pad_wdt_kernel(const float* __restrict__ src,
                               __bf16* __restrict__ dst) {
    int i = blockIdx.x * blockDim.x + threadIdx.x;
    if (i >= DIc * KDT) return;
    int row = i / KDT, col = i % KDT;
    dst[i] = (col < DTRc) ? f2bf(src[row * DTRc + col]) : f2bf(0.f);
}

// dt slice of x_dbl (TOK,80)[:, :48] -> bf16 (TOK,64) zero-padded
__global__ void prep_dt_kernel(const float* __restrict__ x_dbl,
                               __bf16* __restrict__ dt_pad) {
    int i = blockIdx.x * blockDim.x + threadIdx.x;
    if (i >= TOK * KDT) return;
    int row = i / KDT, col = i % KDT;
    dt_pad[i] = (col < DTRc) ? f2bf(x_dbl[row * NXc + col]) : f2bf(0.f);
}

// ---------------------------------------------------------------------------
// LayerNorm: one 256-thread block per token (D=768 -> 3 elems/thread)
// ---------------------------------------------------------------------------
__global__ void layernorm_kernel(const float* __restrict__ x,
                                 const float* __restrict__ g,
                                 const float* __restrict__ b,
                                 __bf16* __restrict__ xn) {
    __shared__ float s1[256], s2[256];
    const int tok = blockIdx.x;
    const int tid = threadIdx.x;
    const float* row = x + (size_t)tok * Dd;

    float a0 = row[tid], a1 = row[tid + 256], a2 = row[tid + 512];
    s1[tid] = a0 + a1 + a2;
    s2[tid] = a0 * a0 + a1 * a1 + a2 * a2;
    __syncthreads();
    for (int off = 128; off > 0; off >>= 1) {
        if (tid < off) { s1[tid] += s1[tid + off]; s2[tid] += s2[tid + off]; }
        __syncthreads();
    }
    const float mu  = s1[0] * (1.f / Dd);
    const float var = s2[0] * (1.f / Dd) - mu * mu;
    const float inv = rsqrtf(var + EPSc);

    __bf16* orow = xn + (size_t)tok * Dd;
    orow[tid]       = f2bf((a0 - mu) * inv * g[tid]       + b[tid]);
    orow[tid + 256] = f2bf((a1 - mu) * inv * g[tid + 256] + b[tid + 256]);
    orow[tid + 512] = f2bf((a2 - mu) * inv * g[tid + 512] + b[tid + 512]);
}

// ---------------------------------------------------------------------------
// Causal depthwise conv1d(width 4) + bias + SiLU over x = xz[:, :DI]
// ---------------------------------------------------------------------------
__global__ void conv_silu_kernel(const float* __restrict__ xz,
                                 const float* __restrict__ Wc,
                                 const float* __restrict__ bc,
                                 float*  __restrict__ xc_f32,
                                 __bf16* __restrict__ xc_bf) {
    int i = blockIdx.x * blockDim.x + threadIdx.x;
    if (i >= TOK * DIc) return;
    const int c   = i % DIc;
    const int tok = i / DIc;
    const int l   = tok % Ll;
    const int bB  = tok / Ll;

    float acc = bc[c];
#pragma unroll
    for (int j = 0; j < 4; ++j) {
        int li = l - 3 + j;
        if (li >= 0)
            acc += Wc[c * 4 + j] * xz[((size_t)(bB * Ll + li)) * (2 * DIc) + c];
    }
    float v = acc / (1.f + expf(-acc));          // SiLU
    xc_f32[i] = v;
    xc_bf[i]  = f2bf(v);
}

// ---------------------------------------------------------------------------
// SSM scan: one thread per (b, d); 16-wide state in registers; fused
// softplus(delta), dA=exp(delta*A), h update, C-contraction, D*u skip,
// SiLU(z) gating. Output y as bf16 for the WMMA out-projection.
// ---------------------------------------------------------------------------
__global__ void ssm_scan_kernel(const float* __restrict__ delta_pre, // (TOK,DI)
                                const float* __restrict__ b_dt,      // (DI)
                                const float* __restrict__ A_log,     // (DI,N)
                                const float* __restrict__ D_param,   // (DI)
                                const float* __restrict__ x_dbl,     // (TOK,80)
                                const float* __restrict__ xc_f32,    // (TOK,DI)
                                const float* __restrict__ xz,        // (TOK,2DI)
                                __bf16* __restrict__ y_bf) {         // (TOK,DI)
    int i = blockIdx.x * blockDim.x + threadIdx.x;
    if (i >= Bb * DIc) return;
    const int d  = i % DIc;
    const int bB = i / DIc;

    float Ad[Nst];
#pragma unroll
    for (int n = 0; n < Nst; ++n) Ad[n] = -expf(A_log[d * Nst + n]);

    const float bdt = b_dt[d];
    const float Dp  = D_param[d];

    float h[Nst];
#pragma unroll
    for (int n = 0; n < Nst; ++n) h[n] = 0.f;

    for (int l = 0; l < Ll; ++l) {
        const size_t tok = (size_t)bB * Ll + l;
        const float delta = softplusf(delta_pre[tok * DIc + d] + bdt);
        const float u     = xc_f32[tok * DIc + d];
        const float zv    = xz[tok * (2 * DIc) + DIc + d];
        const float* bc   = x_dbl + tok * NXc;   // B at +48, C at +64

        const float du = delta * u;
        float y = 0.f;
#pragma unroll
        for (int n = 0; n < Nst; ++n) {
            const float dA = expf(delta * Ad[n]);
            h[n] = dA * h[n] + du * bc[DTRc + n];
            y += h[n] * bc[DTRc + Nst + n];
        }
        y += Dp * u;
        y *= zv / (1.f + expf(-zv));             // * SiLU(z)
        y_bf[tok * DIc + d] = f2bf(y);
    }
}

// ---------------------------------------------------------------------------
// out = relu(out_pre) + input_data
// ---------------------------------------------------------------------------
__global__ void epilogue_kernel(const float* __restrict__ out_pre,
                                const float* __restrict__ inp,
                                float* __restrict__ out) {
    int i = blockIdx.x * blockDim.x + threadIdx.x;
    if (i < TOK * Dd) out[i] = fmaxf(out_pre[i], 0.f) + inp[i];
}

// ---------------------------------------------------------------------------
extern "C" void kernel_launch(void* const* d_in, const int* in_sizes, int n_in,
                              void* d_out, int out_size, void* d_ws, size_t ws_size,
                              hipStream_t stream) {
    const float* input   = (const float*)d_in[0];
    const float* ln_g    = (const float*)d_in[1];
    const float* ln_b    = (const float*)d_in[2];
    const float* W_in    = (const float*)d_in[3];
    const float* W_conv  = (const float*)d_in[4];
    const float* b_conv  = (const float*)d_in[5];
    const float* W_x     = (const float*)d_in[6];
    const float* W_dt    = (const float*)d_in[7];
    const float* b_dt    = (const float*)d_in[8];
    const float* A_log   = (const float*)d_in[9];
    const float* D_param = (const float*)d_in[10];
    const float* W_out   = (const float*)d_in[11];
    float* out = (float*)d_out;

    // ---- workspace carve-up (256B aligned) --------------------------------
    char* ws = (char*)d_ws;
    size_t off = 0;
    auto carve = [&](size_t bytes) -> char* {
        char* p = ws + off;
        off = (off + bytes + 255) & ~(size_t)255;
        return p;
    };
    __bf16* xn_bf    = (__bf16*)carve((size_t)TOK * Dd * 2);
    __bf16* Win_bf   = (__bf16*)carve((size_t)2 * DIc * Dd * 2);
    __bf16* Wx_bf    = (__bf16*)carve((size_t)NXc * DIc * 2);
    __bf16* Wdt_bf   = (__bf16*)carve((size_t)DIc * KDT * 2);
    __bf16* Wout_bf  = (__bf16*)carve((size_t)Dd * DIc * 2);
    float*  xz       = (float*) carve((size_t)TOK * 2 * DIc * 4);
    float*  xc_f32   = (float*) carve((size_t)TOK * DIc * 4);
    __bf16* xc_bf    = (__bf16*)carve((size_t)TOK * DIc * 2);
    float*  x_dbl    = (float*) carve((size_t)TOK * NXc * 4);
    __bf16* dt_pad   = (__bf16*)carve((size_t)TOK * KDT * 2);
    float*  delta_pre= (float*) carve((size_t)TOK * DIc * 4);
    __bf16* y_bf     = (__bf16*)carve((size_t)TOK * DIc * 2);
    float*  out_pre  = delta_pre;   // reuse: scan fully consumes delta_pre first

    const int T = 256;
    auto cdiv = [](int a, int b) { return (a + b - 1) / b; };

    // ---- weight conversion ------------------------------------------------
    { int n = 2 * DIc * Dd;
      cvt_f32_bf16_kernel<<<cdiv(n, T), T, 0, stream>>>(W_in, Win_bf, n); }
    { int n = NXc * DIc;
      cvt_f32_bf16_kernel<<<cdiv(n, T), T, 0, stream>>>(W_x, Wx_bf, n); }
    { int n = Dd * DIc;
      cvt_f32_bf16_kernel<<<cdiv(n, T), T, 0, stream>>>(W_out, Wout_bf, n); }
    pad_wdt_kernel<<<cdiv(DIc * KDT, T), T, 0, stream>>>(W_dt, Wdt_bf);

    // ---- LayerNorm --------------------------------------------------------
    layernorm_kernel<<<TOK, 256, 0, stream>>>(input, ln_g, ln_b, xn_bf);

    // ---- GEMM1: xz = xn @ W_in^T   (4096x768)*(3072x768)^T ---------------
    { int waves = (TOK / 32) * (2 * DIc / 64);     // MT=2, NT=4
      gemm_bf16_wmma_kernel<2, 4><<<cdiv(waves, 8), 256, 0, stream>>>(
          xn_bf, Dd, Win_bf, Dd, xz, 2 * DIc, TOK, 2 * DIc, Dd); }

    // ---- depthwise conv + SiLU -------------------------------------------
    conv_silu_kernel<<<cdiv(TOK * DIc, T), T, 0, stream>>>(
        xz, W_conv, b_conv, xc_f32, xc_bf);

    // ---- GEMM2: x_dbl = xc @ W_x^T (4096x1536)*(80x1536)^T ---------------
    { int waves = (TOK / 32) * (NXc / 80);         // MT=2, NT=5 (full N strip)
      gemm_bf16_wmma_kernel<2, 5><<<cdiv(waves, 8), 256, 0, stream>>>(
          xc_bf, DIc, Wx_bf, DIc, x_dbl, NXc, TOK, NXc, DIc); }

    // ---- GEMM3: delta_pre = dt @ W_dt^T (K padded 48->64) ----------------
    prep_dt_kernel<<<cdiv(TOK * KDT, T), T, 0, stream>>>(x_dbl, dt_pad);
    { int waves = (TOK / 32) * (DIc / 64);         // MT=2, NT=4
      gemm_bf16_wmma_kernel<2, 4><<<cdiv(waves, 8), 256, 0, stream>>>(
          dt_pad, KDT, Wdt_bf, KDT, delta_pre, DIc, TOK, DIc, KDT); }

    // ---- SSM scan + gating ------------------------------------------------
    ssm_scan_kernel<<<cdiv(Bb * DIc, T), T, 0, stream>>>(
        delta_pre, b_dt, A_log, D_param, x_dbl, xc_f32, xz, y_bf);

    // ---- GEMM4: out_pre = y @ W_out^T (4096x1536)*(768x1536)^T -----------
    { int waves = (TOK / 32) * (Dd / 64);          // MT=2, NT=4
      gemm_bf16_wmma_kernel<2, 4><<<cdiv(waves, 8), 256, 0, stream>>>(
          y_bf, DIc, Wout_bf, DIc, out_pre, Dd, TOK, Dd, DIc); }

    // ---- ReLU + residual --------------------------------------------------
    epilogue_kernel<<<cdiv(TOK * Dd, T), T, 0, stream>>>(out_pre, input, out);
}
